// PseudoSparseConv3d_57397942944140
// MI455X (gfx1250) — matrix operand order repaired and checked by reading, not compiled, not run on hardware
//
#include <hip/hip_runtime.h>

// Problem constants (match reference)
#define D_   32
#define H_   256
#define W_   256
#define V_   40000
#define CIN  16
#define COUT 32
#define P_   32
#define DHW  (D_*H_*W_)
#define EPSV 1e-5f

typedef _Float16 half16 __attribute__((ext_vector_type(16)));
typedef float    float8 __attribute__((ext_vector_type(8)));
typedef float    fvec4  __attribute__((ext_vector_type(4)));

// Workspace layout (bytes). Requires ws_size >= 64 MB + 64 KB.
//   [0, 28672)          : packed A tiles (2 Mblk x 14 Kblk x 32 lanes x 16 halves)
//   [32768, 32896)      : per-channel sum   (32 f32)
//   [32896, 33024)      : per-channel sumsq (32 f32)
//   [33024, 33152)      : BN scale (32 f32)
//   [33152, 33280)      : BN shift (32 f32)
//   [65536, 65536+64MB) : f16 grid, channel-last [d][h][w][ci]
#define WS_A_OFF     0
#define WS_SUM_OFF   32768
#define WS_SQ_OFF    (32768 + 128)
#define WS_SCALE_OFF (32768 + 256)
#define WS_SHIFT_OFF (32768 + 384)
#define WS_GRID_OFF  65536

// ---------------------------------------------------------------------------
// Pack conv weights (COUT,CIN,3,3,3) f32 into a per-lane WMMA A layout (contiguous k, same as B)
// (16-bit A 16x32): lane<16 holds M=lane, half t -> K = t;
// lane>=16 holds M=lane-16, half t -> K = t+16.
// GEMM K ordering: k = kidx*16 + ci, kidx = kd*9+kh*3+kw, padded 432->448.
// ---------------------------------------------------------------------------
__global__ void pack_w(const float* __restrict__ cw, _Float16* __restrict__ wsA) {
    int blk  = blockIdx.x;          // 0..27 = mb*14 + jb
    int lane = threadIdx.x;         // 0..31
    int mb = blk / 14, jb = blk % 14;
    int m  = lane & 15;
    int hi = lane >> 4;
#pragma unroll
    for (int t = 0; t < 16; ++t) {
        int klocal = (hi ? 16 : 0) + t;
        int k = jb * 32 + klocal;
        int kidx = k >> 4, ci = k & 15;
        float v = 0.f;
        if (kidx < 27) v = cw[((mb * 16 + m) * CIN + ci) * 27 + kidx];
        wsA[((size_t)(blk * 32 + lane)) * 16 + t] = (_Float16)v;
    }
}

// ---------------------------------------------------------------------------
// Mean over P=32 and scatter into f16 channel-last grid.
// One thread per (voxel, channel): reads 128 contiguous bytes, writes 2 bytes
// (16 adjacent channel threads write 16 coalesced f16 = one 32B group).
// ---------------------------------------------------------------------------
__global__ void scatter_mean(const float* __restrict__ vox,
                             const int*   __restrict__ idx,
                             _Float16*    __restrict__ grid) {
    int t = blockIdx.x * blockDim.x + threadIdx.x;
    if (t >= V_ * CIN) return;
    int v = t >> 4, ci = t & 15;
    int w = idx[3 * v + 0], h = idx[3 * v + 1], d = idx[3 * v + 2];
    if ((unsigned)w >= W_ || (unsigned)h >= H_ || (unsigned)d >= D_) return;
    const fvec4* p4 = (const fvec4*)(vox + ((size_t)v * CIN + ci) * P_);
    float s = 0.f;
#pragma unroll
    for (int i = 0; i < 8; ++i) { fvec4 q = p4[i]; s += q.x + q.y + q.z + q.w; }
    grid[(((size_t)d * H_ + h) * W_ + w) * CIN + ci] = (_Float16)(s * (1.0f / P_));
}

// ---------------------------------------------------------------------------
// Implicit-GEMM conv via v_wmma_f32_16x16x32_f16.
// Block = 256 threads = 8 waves = 4 N-tiles (16 w each) x 2 M-blocks at one (d,h).
// Per wave per K-block: one 32B contiguous B load per lane (channel-last grid),
// one LDS A load, one WMMA. Fully unrolled: 14 WMMAs per wave, loads clustered
// so the scheduler can clause b128 pairs and overlap with the WMMA chain.
// Per-channel sum / sumsq for BatchNorm reduced with shfl_xor butterflies
// (conflict-free ds ops) before touching LDS atomics.
// ---------------------------------------------------------------------------
__global__ void __launch_bounds__(256)
conv_wmma(const _Float16* __restrict__ grid,
          const _Float16* __restrict__ wsA,
          const float*    __restrict__ bias,
          float*          __restrict__ out,
          float*          __restrict__ gsum,
          float*          __restrict__ gsq) {
    __shared__ _Float16 sA[28 * 32 * 16];   // 28 KB: all packed A tiles
    __shared__ float ssum[COUT], ssq[COUT];

    int tid = threadIdx.x;
    if (tid < COUT) { ssum[tid] = 0.f; ssq[tid] = 0.f; }
    {   // cooperative A stage: 28672 B = 7168 dwords, coalesced across threads
        const uint32_t* src = (const uint32_t*)wsA;
        uint32_t*       dst = (uint32_t*)sA;
#pragma unroll
        for (int i = 0; i < 28 * 32 * 16 / 2 / 256; ++i)
            dst[tid + i * 256] = src[tid + i * 256];
    }
    __syncthreads();

    int wave = tid >> 5, lane = tid & 31;
    int mb = wave & 1, nt = wave >> 1;
    int d = blockIdx.z, h = blockIdx.y;
    int w0 = blockIdx.x * 64 + nt * 16;
    int n = lane & 15, hi = lane >> 4;

    float8 acc;
#pragma unroll
    for (int r = 0; r < 8; ++r) acc[r] = 0.f;

#pragma unroll
    for (int jb = 0; jb < 14; ++jb) {
        int kidx = 2 * jb + hi;                   // this lane-half's kernel offset
        int kd = kidx / 9, kh = (kidx % 9) / 3, kw = kidx % 3;
        int dd = d + kd - 1, hh = h + kh - 1, ww = w0 + n + kw - 1;
        bool valid = (kidx < 27) && (unsigned)dd < D_ &&
                     (unsigned)hh < H_ && (unsigned)ww < W_;
        half16 b;
#pragma unroll
        for (int i = 0; i < 16; ++i) b[i] = (_Float16)0.f;
        if (valid)
            b = *(const half16*)(grid + (((size_t)dd * H_ + hh) * W_ + ww) * CIN);
        half16 a = *(const half16*)&sA[((mb * 14 + jb) * 32 + lane) * 16];
        acc = __builtin_amdgcn_wmma_f32_16x16x32_f16(
            /*neg_a=*/false, a, /*neg_b=*/false, b,
            /*c_mod=*/(short)0, acc, /*reuse_a=*/false, /*reuse_b=*/false);
    }

    // C layout: VGPR r, lanes 0-15 -> M=r, lanes 16-31 -> M=r+8; N = lane%16.
    int w = w0 + n;
#pragma unroll
    for (int r = 0; r < 8; ++r) {
        int co = mb * 16 + hi * 8 + r;
        float v = acc[r] + bias[co];
        out[(((size_t)co * D_ + d) * H_ + h) * W_ + w] = v;
        // Butterfly-reduce across the 16 lanes that share channel `co`
        // (xor over the N sub-lane bits) -> conflict-free ds permutes,
        // then a single LDS atomic per (r, half-wave).
        float s = v, q = v * v;
#pragma unroll
        for (int mask = 8; mask >= 1; mask >>= 1) {
            s += __shfl_xor(s, mask, 32);
            q += __shfl_xor(q, mask, 32);
        }
        if (n == 0) { atomicAdd(&ssum[co], s); atomicAdd(&ssq[co], q); }
    }
    __syncthreads();
    if (tid < COUT) { atomicAdd(&gsum[tid], ssum[tid]); atomicAdd(&gsq[tid], ssq[tid]); }
}

// ---------------------------------------------------------------------------
// Fold sums into affine normalize coefficients.
// ---------------------------------------------------------------------------
__global__ void finalize_stats(const float* __restrict__ gsum,
                               const float* __restrict__ gsq,
                               const float* __restrict__ gamma,
                               const float* __restrict__ beta,
                               float* __restrict__ scale,
                               float* __restrict__ shift) {
    int c = threadIdx.x;
    if (c >= COUT) return;
    const float invN = 1.0f / (float)DHW;
    float mean = gsum[c] * invN;
    float var  = gsq[c] * invN - mean * mean;
    float inv  = rsqrtf(var + EPSV);
    float sc   = gamma[c] * inv;
    scale[c] = sc;
    shift[c] = beta[c] - mean * sc;
}

// ---------------------------------------------------------------------------
// In-place BN + ReLU over d_out, vectorized 16B (DHW % 4 == 0 so each
// fvec4 stays inside one channel). Pure streaming pass: non-temporal
// load + non-temporal store so it does not rinse L2.
// ---------------------------------------------------------------------------
__global__ void __launch_bounds__(256)
bn_relu(float* __restrict__ out,
        const float* __restrict__ scale, const float* __restrict__ shift) {
    size_t i4 = (size_t)blockIdx.x * 256 + threadIdx.x;
    int c = (int)((i4 * 4) / DHW);
    fvec4* p = (fvec4*)out;
    fvec4 x = __builtin_nontemporal_load(&p[i4]);
    float sc = scale[c], sh = shift[c];
    x.x = fmaxf(fmaf(x.x, sc, sh), 0.f);
    x.y = fmaxf(fmaf(x.y, sc, sh), 0.f);
    x.z = fmaxf(fmaf(x.z, sc, sh), 0.f);
    x.w = fmaxf(fmaf(x.w, sc, sh), 0.f);
    __builtin_nontemporal_store(x, &p[i4]);
}

extern "C" void kernel_launch(void* const* d_in, const int* in_sizes, int n_in,
                              void* d_out, int out_size, void* d_ws, size_t ws_size,
                              hipStream_t stream) {
    (void)in_sizes; (void)n_in; (void)out_size; (void)ws_size;
    const float* voxels  = (const float*)d_in[0];
    const int*   indices = (const int*)d_in[1];
    const float* conv_w  = (const float*)d_in[2];
    const float* conv_b  = (const float*)d_in[3];
    const float* gamma   = (const float*)d_in[4];
    const float* beta    = (const float*)d_in[5];

    char* ws = (char*)d_ws;
    _Float16* wsA   = (_Float16*)(ws + WS_A_OFF);
    float*    gsum  = (float*)(ws + WS_SUM_OFF);
    float*    gsq   = (float*)(ws + WS_SQ_OFF);
    float*    scale = (float*)(ws + WS_SCALE_OFF);
    float*    shift = (float*)(ws + WS_SHIFT_OFF);
    _Float16* grid  = (_Float16*)(ws + WS_GRID_OFF);
    float*    out   = (float*)d_out;

    // Zero f16 grid (f16 zero == 0x0000) and BN accumulators each call.
    (void)hipMemsetAsync(grid, 0, (size_t)DHW * CIN * sizeof(_Float16), stream);
    (void)hipMemsetAsync(gsum, 0, 256, stream);   // covers gsum + gsq

    pack_w<<<28, 32, 0, stream>>>(conv_w, wsA);
    scatter_mean<<<(V_ * CIN + 255) / 256, 256, 0, stream>>>(voxels, indices, grid);
    conv_wmma<<<dim3(W_ / 64, H_, D_), 256, 0, stream>>>(grid, wsA, conv_b, out, gsum, gsq);
    finalize_stats<<<1, 32, 0, stream>>>(gsum, gsq, gamma, beta, scale, shift);
    bn_relu<<<(unsigned)(((size_t)COUT * DHW / 4 + 255) / 256), 256, 0, stream>>>(out, scale, shift);
}
